// SelfAttention_58849641889886
// MI455X (gfx1250) — matrix operand order repaired and checked by reading, not compiled, not run on hardware
//
#include <hip/hip_runtime.h>

// ---------------------------------------------------------------------------
// Attention block for MI455X (gfx1250, wave32, WMMA f16->f32 + TDM).
//   B=2, C=512, H=W=64 (N=4096), heads=8, hd=64, groups=32.
// All WMMA operand tiles are DMA'd global->LDS by the Tensor Data Mover
// (tensor_load_to_lds, TENSORcnt, double-buffered), with the D# pad feature
// producing the 72-half LDS row stride. Fragment loads are contiguous b128
// pairs: A tiles [m][k], B tiles [n][k] (N-major).
// Global layouts: h/attn-out [B][N][C], Q,K [bh][n][d], Vt [bh][d][n].
// ---------------------------------------------------------------------------

typedef __attribute__((ext_vector_type(16))) _Float16 v16h;
typedef __attribute__((ext_vector_type(8)))  _Float16 v8h;
typedef __attribute__((ext_vector_type(8)))  float    v8f;
typedef __attribute__((ext_vector_type(4)))  unsigned int u32x4;
typedef __attribute__((ext_vector_type(8)))  int      i32x8;
typedef __attribute__((ext_vector_type(4)))  int      i32x4;

#define C_DIM 512
#define N_DIM 4096
#define NHEAD 8
#define HDIM  64
#define LDT   72   // LDS row stride in halfs (64 data + 8 pad, made by TDM pad)

// ---- TDM: DMA one [tile_h x 64-half] tile (row stride in elems) into LDS ---
// D# per CDNA5 ISA 8.3-8.5: data_size=2B, pad_enable, pad_interval=4 (128B),
// pad_amount=3 (16B) -> LDS row stride 144B = 72 halfs. tensor dims = tile
// dims so no OOB path is taken. Groups 2/3 unused (2D tensor).
// This toolchain exposes the 6-arg builtin: (g0, g1, g2, g3, g4, cpol).
__device__ inline void tdm_load_tile(const _Float16* lds_dst, const _Float16* gsrc,
                                     unsigned row_stride, unsigned tile_h) {
    const unsigned long long ga = (unsigned long long)(uintptr_t)gsrc;
    u32x4 g0 = {};
    g0[0] = 1u;                                             // count=1, user D#
    g0[1] = (unsigned)(uintptr_t)lds_dst;                   // lds_addr (bytes)
    g0[2] = (unsigned)ga;                                   // global_addr[31:0]
    g0[3] = ((unsigned)(ga >> 32) & 0x01FFFFFFu) | (2u << 30);  // addr[56:32], type=2
    i32x8 g1 = {};
    g1[0] = (int)((1u << 16) | (1u << 20) | (4u << 22) | (3u << 25));
    g1[1] = (int)(64u << 16);                               // tensor_dim0 = 64
    g1[2] = (int)((tile_h & 0xFFFFu) << 16);                // tensor_dim1 = tile_h
    g1[3] = (int)(64u << 16);                               // tile_dim0 = 64
    g1[4] = (int)(tile_h & 0xFFFFu);                        // tile_dim1 = tile_h
    g1[5] = (int)row_stride;                                // tensor_dim0_stride
    const i32x4 z4 = {};
    const i32x8 z8 = {};
    __builtin_amdgcn_tensor_load_to_lds(g0, g1, z4, z4, z8, 0);
}

// A (16x32 f16, MxK): lane L -> row m = L&15, hf = L>>4. Two b128 loads.
__device__ inline v16h load_a_frag(const _Float16* __restrict__ s, int ld) {
    const int lane = threadIdx.x & 31;
    const int m = lane & 15, hf = lane >> 4;
    union { v16h v; v8h h[2]; } u;
    u.h[0] = *(const v8h*)&s[m * ld + 8 * hf];
    u.h[1] = *(const v8h*)&s[m * ld + 16 + 8 * hf];
    return u.v;
}
// B (32x16 f16, KxN) from N-major staging Bt[n][k]. Two b128 loads.
__device__ inline v16h load_bT_frag(const _Float16* __restrict__ s, int ld) {
    const int lane = threadIdx.x & 31;
    const int n = lane & 15, hf = lane >> 4;
    union { v16h v; v8h h[2]; } u;
    u.h[0] = *(const v8h*)&s[n * ld + 8 * hf];
    u.h[1] = *(const v8h*)&s[n * ld + 16 + 8 * hf];
    return u.v;
}

__device__ inline float wave_sum(float v) {
#pragma unroll
    for (int m = 16; m >= 1; m >>= 1) v += __shfl_xor(v, m, 32);
    return v;
}

// ---- Kernel 1: GroupNorm -> f16, pixel-major hT[B][N][C] -------------------
__global__ void gn_kernel(const float* __restrict__ x, const float* __restrict__ w,
                          const float* __restrict__ bias, _Float16* __restrict__ hT) {
    const int tid = threadIdx.x;
    const int b = blockIdx.x >> 5, g = blockIdx.x & 31;
    const float* xp = x + ((size_t)b * C_DIM + g * 16) * N_DIM;  // 16 ch x 4096 px
    float s = 0.f, s2 = 0.f;
    for (int i = tid; i < 65536; i += 256) { float v = xp[i]; s += v; s2 += v * v; }
    s = wave_sum(s); s2 = wave_sum(s2);
    __shared__ float rs[8], rs2[8], stats[2], ws_[16], bs_[16];
    const int lane = tid & 31, wid = tid >> 5;
    if (lane == 0) { rs[wid] = s; rs2[wid] = s2; }
    if (tid < 16) { ws_[tid] = w[g * 16 + tid]; bs_[tid] = bias[g * 16 + tid]; }
    __syncthreads();
    if (tid == 0) {
        float S = 0.f, S2 = 0.f;
        for (int i = 0; i < 8; ++i) { S += rs[i]; S2 += rs2[i]; }
        const float mean = S * (1.f / 65536.f);
        const float var  = S2 * (1.f / 65536.f) - mean * mean;
        stats[0] = mean; stats[1] = rsqrtf(var + 1e-5f);
    }
    __syncthreads();
    const float mean = stats[0], rstd = stats[1];
    for (int n = tid; n < N_DIM; n += 256) {
        alignas(16) _Float16 tmp[16];
#pragma unroll
        for (int j = 0; j < 16; ++j)
            tmp[j] = (_Float16)((xp[(size_t)j * N_DIM + n] - mean) * rstd * ws_[j] + bs_[j]);
        _Float16* dst = hT + ((size_t)b * N_DIM + n) * C_DIM + g * 16;
        *(v8h*)dst       = *(const v8h*)tmp;
        *(v8h*)(dst + 8) = *(const v8h*)(tmp + 8);
    }
}

// ---- Kernel 2: f32 -> f16 conversion --------------------------------------
__global__ void cvt_kernel(const float* __restrict__ src, _Float16* __restrict__ dst, int n) {
    const int i = blockIdx.x * blockDim.x + threadIdx.x;
    if (i < n) dst[i] = (_Float16)src[i];
}

// ---- Kernel 3: QKV GEMM  Out[1536,4096] = Wqkv[1536,512] x h ---------------
// grid (32, 24, 2), block 128 (4 waves). 64(m) x 128(n) tile, TDM double-buffer.
__global__ void qkv_gemm(const _Float16* __restrict__ W, const _Float16* __restrict__ HT,
                         _Float16* __restrict__ Q, _Float16* __restrict__ K,
                         _Float16* __restrict__ Vt) {
    __shared__ __align__(16) _Float16 Ws[2][64 * LDT];     // [m][k]
    __shared__ __align__(16) _Float16 Ht[2][128 * LDT];    // [n][k]
    const int tid = threadIdx.x, lane = tid & 31, wid = tid >> 5;
    const int n0 = blockIdx.x * 128;
    const int m0 = blockIdx.y * 64;
    const int b  = blockIdx.z;
    const _Float16* Wb = W + (size_t)m0 * C_DIM;
    const _Float16* Hb = HT + (size_t)b * N_DIM * C_DIM + (size_t)n0 * C_DIM;
    v8f acc[8] = {};
    if (wid == 0) {                                        // prefetch chunk 0
        tdm_load_tile(Ws[0], Wb, C_DIM, 64);
        tdm_load_tile(Ht[0], Hb, C_DIM, 128);
    }
    for (int kc = 0; kc < 8; ++kc) {
        const int cur = kc & 1;
        if (wid == 0) {
            if (kc < 7) {
                tdm_load_tile(Ws[cur ^ 1], Wb + (kc + 1) * 64, C_DIM, 64);
                tdm_load_tile(Ht[cur ^ 1], Hb + (kc + 1) * 64, C_DIM, 128);
                __builtin_amdgcn_s_wait_tensorcnt(2);      // current pair done
            } else {
                __builtin_amdgcn_s_wait_tensorcnt(0);
            }
        }
        __syncthreads();
        const _Float16* Wt_ = Ws[cur];
        const _Float16* Ht_ = Ht[cur];
#pragma unroll
        for (int kk = 0; kk < 2; ++kk) {
            const v16h a = load_a_frag(&Wt_[(wid * 16) * LDT + kk * 32], LDT);
            v16h bf[8];
#pragma unroll
            for (int nt = 0; nt < 8; ++nt)
                bf[nt] = load_bT_frag(&Ht_[(nt * 16) * LDT + kk * 32], LDT);
#pragma unroll
            for (int nt = 0; nt < 8; ++nt)
                acc[nt] = __builtin_amdgcn_wmma_f32_16x16x32_f16(
                    false, a, false, bf[nt], (short)0, acc[nt], false, false);
        }
        __syncthreads();
    }
    const int hf = lane >> 4;
    const int obase = m0 + wid * 16 + 8 * hf;              // 8-aligned out channel base
    const int which = obase >> 9;                          // 0=q 1=k 2=v (uniform)
    const int cb = obase & 511, hh = cb >> 6, dbase = cb & 63;
    const size_t bh = (size_t)(b * NHEAD + hh);
    const float oscale = (which == 0) ? 0.125f : 1.f;      // fold hd^-0.5 into Q
#pragma unroll
    for (int nt = 0; nt < 8; ++nt) {
        const int n = n0 + nt * 16 + (lane & 15);
        alignas(16) _Float16 pk[8];
#pragma unroll
        for (int r = 0; r < 8; ++r) pk[r] = (_Float16)(acc[nt][r] * oscale);
        if (which == 0)      *(v8h*)&Q[(bh * N_DIM + n) * HDIM + dbase] = *(const v8h*)pk;
        else if (which == 1) *(v8h*)&K[(bh * N_DIM + n) * HDIM + dbase] = *(const v8h*)pk;
        else {
#pragma unroll
            for (int r = 0; r < 8; ++r)
                Vt[(bh * HDIM + dbase + r) * N_DIM + n] = pk[r];
        }
    }
}

// ---- Kernel 4: flash attention, 64-query blocks, online softmax ------------
// grid (64, 16), block 128 (4 waves). TDM double-buffered K/V. Out [B][N][C].
__global__ void flash_attn(const _Float16* __restrict__ Q, const _Float16* __restrict__ K,
                           const _Float16* __restrict__ Vt, _Float16* __restrict__ AttT) {
    __shared__ __align__(16) _Float16 Qs[64 * LDT];        // [q][d]   (A for S)
    __shared__ __align__(16) _Float16 Ks[2][64 * LDT];     // [key][d] (B for S)
    __shared__ __align__(16) _Float16 Vs[2][64 * LDT];     // [d][key] (B for O)
    __shared__ __align__(16) _Float16 Ps[64 * LDT];        // [q][key] (A for O)
    const int tid = threadIdx.x, lane = tid & 31, wid = tid >> 5;
    const int q0 = blockIdx.x * 64;
    const int bh = blockIdx.y;
    const _Float16* Qb  = Q  + (size_t)bh * N_DIM * HDIM;
    const _Float16* Kb  = K  + (size_t)bh * N_DIM * HDIM;
    const _Float16* Vtb = Vt + (size_t)bh * HDIM * N_DIM;
    if (wid == 0) {
        tdm_load_tile(Qs,    Qb + (size_t)q0 * HDIM, HDIM, 64);
        tdm_load_tile(Ks[0], Kb,                      HDIM, 64);
        tdm_load_tile(Vs[0], Vtb,                     N_DIM, 64);
        __builtin_amdgcn_s_wait_tensorcnt(2);              // Qs ready (in-order)
    }
    __syncthreads();
    v16h aQ[2];
    aQ[0] = load_a_frag(&Qs[(wid * 16) * LDT +  0], LDT);
    aQ[1] = load_a_frag(&Qs[(wid * 16) * LDT + 32], LDT);

    float m_i[8], l_i[8];
#pragma unroll
    for (int r = 0; r < 8; ++r) { m_i[r] = -3.0e38f; l_i[r] = 0.f; }
    v8f accO[4] = {};
    const int hf = lane >> 4;

    for (int it = 0; it < 64; ++it) {
        const int cur = it & 1;
        if (wid == 0) {
            if (it < 63) {
                tdm_load_tile(Ks[cur ^ 1], Kb + (size_t)(it + 1) * 64 * HDIM, HDIM, 64);
                tdm_load_tile(Vs[cur ^ 1], Vtb + (it + 1) * 64,               N_DIM, 64);
                __builtin_amdgcn_s_wait_tensorcnt(2);
            } else {
                __builtin_amdgcn_s_wait_tensorcnt(0);
            }
        }
        __syncthreads();
        const _Float16* Ks_ = Ks[cur];
        const _Float16* Vs_ = Vs[cur];
        // S = Q K^T : rows = queries, cols = keys
        v8f s[4] = {};
#pragma unroll
        for (int kk = 0; kk < 2; ++kk) {
            v16h bf[4];
#pragma unroll
            for (int nt = 0; nt < 4; ++nt)
                bf[nt] = load_bT_frag(&Ks_[(nt * 16) * LDT + kk * 32], LDT);
#pragma unroll
            for (int nt = 0; nt < 4; ++nt)
                s[nt] = __builtin_amdgcn_wmma_f32_16x16x32_f16(
                    false, aQ[kk], false, bf[nt], (short)0, s[nt], false, false);
        }
        // online softmax (row spans the 16-lane column group)
        float rmax[8];
#pragma unroll
        for (int r = 0; r < 8; ++r)
            rmax[r] = fmaxf(fmaxf(s[0][r], s[1][r]), fmaxf(s[2][r], s[3][r]));
#pragma unroll
        for (int msk = 1; msk < 16; msk <<= 1)
#pragma unroll
            for (int r = 0; r < 8; ++r) rmax[r] = fmaxf(rmax[r], __shfl_xor(rmax[r], msk, 32));
        float sc[8], rsum[8];
#pragma unroll
        for (int r = 0; r < 8; ++r) {
            const float mn = fmaxf(m_i[r], rmax[r]);
            sc[r] = __expf(m_i[r] - mn);
            m_i[r] = mn; rsum[r] = 0.f;
        }
#pragma unroll
        for (int nt = 0; nt < 4; ++nt)
#pragma unroll
            for (int r = 0; r < 8; ++r) {
                const float p = __expf(s[nt][r] - m_i[r]);
                s[nt][r] = p; rsum[r] += p;
            }
#pragma unroll
        for (int msk = 1; msk < 16; msk <<= 1)
#pragma unroll
            for (int r = 0; r < 8; ++r) rsum[r] += __shfl_xor(rsum[r], msk, 32);
#pragma unroll
        for (int r = 0; r < 8; ++r) l_i[r] = l_i[r] * sc[r] + rsum[r];
#pragma unroll
        for (int nt = 0; nt < 4; ++nt)
#pragma unroll
            for (int r = 0; r < 8; ++r) accO[nt][r] *= sc[r];
        // restage P (C/D layout -> A layout) via wave-private LDS rows
#pragma unroll
        for (int nt = 0; nt < 4; ++nt)
#pragma unroll
            for (int r = 0; r < 8; ++r)
                Ps[(wid * 16 + r + 8 * hf) * LDT + nt * 16 + (lane & 15)] = (_Float16)s[nt][r];
        asm volatile("" ::: "memory");   // keep LDS write->read ordered (in-order DS hw)
        const v16h aP0 = load_a_frag(&Ps[(wid * 16) * LDT +  0], LDT);
        const v16h aP1 = load_a_frag(&Ps[(wid * 16) * LDT + 32], LDT);
        v16h bv[4][2];
#pragma unroll
        for (int dt = 0; dt < 4; ++dt) {
            bv[dt][0] = load_bT_frag(&Vs_[(dt * 16) * LDT +  0], LDT);
            bv[dt][1] = load_bT_frag(&Vs_[(dt * 16) * LDT + 32], LDT);
        }
#pragma unroll
        for (int dt = 0; dt < 4; ++dt) {
            accO[dt] = __builtin_amdgcn_wmma_f32_16x16x32_f16(
                false, aP0, false, bv[dt][0], (short)0, accO[dt], false, false);
            accO[dt] = __builtin_amdgcn_wmma_f32_16x16x32_f16(
                false, aP1, false, bv[dt][1], (short)0, accO[dt], false, false);
        }
        __syncthreads();
    }
    // finalize: /l, write [B][N][C] (16 lanes -> 32 contiguous bytes)
    const int b = bh >> 3, hh = bh & 7;
#pragma unroll
    for (int dt = 0; dt < 4; ++dt) {
        const int d = dt * 16 + (lane & 15);
#pragma unroll
        for (int r = 0; r < 8; ++r) {
            const int n = q0 + wid * 16 + r + 8 * hf;
            AttT[((size_t)b * N_DIM + n) * C_DIM + hh * HDIM + d] =
                (_Float16)(accO[dt][r] / l_i[r]);
        }
    }
}

// ---- Kernel 5: proj GEMM + bias + residual -> f32 out ----------------------
// grid (32, 8, 2), block 128. 64(m) x 128(n) tile, TDM double-buffer.
__global__ void proj_gemm(const _Float16* __restrict__ Wp, const _Float16* __restrict__ AT,
                          const float* __restrict__ bias, const float* __restrict__ x,
                          float* __restrict__ out) {
    __shared__ __align__(16) _Float16 Ws[2][64 * LDT];     // [m][k]
    __shared__ __align__(16) _Float16 At[2][128 * LDT];    // [n][k]
    const int tid = threadIdx.x, lane = tid & 31, wid = tid >> 5;
    const int n0 = blockIdx.x * 128;
    const int m0 = blockIdx.y * 64;
    const int b  = blockIdx.z;
    const _Float16* Wb = Wp + (size_t)m0 * C_DIM;
    const _Float16* Ab = AT + (size_t)b * N_DIM * C_DIM + (size_t)n0 * C_DIM;
    v8f acc[8] = {};
    if (wid == 0) {
        tdm_load_tile(Ws[0], Wb, C_DIM, 64);
        tdm_load_tile(At[0], Ab, C_DIM, 128);
    }
    for (int kc = 0; kc < 8; ++kc) {
        const int cur = kc & 1;
        if (wid == 0) {
            if (kc < 7) {
                tdm_load_tile(Ws[cur ^ 1], Wb + (kc + 1) * 64, C_DIM, 64);
                tdm_load_tile(At[cur ^ 1], Ab + (kc + 1) * 64, C_DIM, 128);
                __builtin_amdgcn_s_wait_tensorcnt(2);
            } else {
                __builtin_amdgcn_s_wait_tensorcnt(0);
            }
        }
        __syncthreads();
        const _Float16* Wt_ = Ws[cur];
        const _Float16* At_ = At[cur];
#pragma unroll
        for (int kk = 0; kk < 2; ++kk) {
            const v16h a = load_a_frag(&Wt_[(wid * 16) * LDT + kk * 32], LDT);
            v16h bf[8];
#pragma unroll
            for (int nt = 0; nt < 8; ++nt)
                bf[nt] = load_bT_frag(&At_[(nt * 16) * LDT + kk * 32], LDT);
#pragma unroll
            for (int nt = 0; nt < 8; ++nt)
                acc[nt] = __builtin_amdgcn_wmma_f32_16x16x32_f16(
                    false, a, false, bf[nt], (short)0, acc[nt], false, false);
        }
        __syncthreads();
    }
    const int hf = lane >> 4;
#pragma unroll
    for (int nt = 0; nt < 8; ++nt) {
        const int n = n0 + nt * 16 + (lane & 15);
#pragma unroll
        for (int r = 0; r < 8; ++r) {
            const int o = m0 + wid * 16 + r + 8 * hf;
            const size_t idx = ((size_t)b * C_DIM + o) * N_DIM + n;
            out[idx] = x[idx] + bias[o] + acc[nt][r];
        }
    }
}

// ---------------------------------------------------------------------------
extern "C" void kernel_launch(void* const* d_in, const int* in_sizes, int n_in,
                              void* d_out, int out_size, void* d_ws, size_t ws_size,
                              hipStream_t stream) {
    const float* x     = (const float*)d_in[0];   // [2,512,64,64]
    const float* gnw   = (const float*)d_in[1];   // [512]
    const float* gnb   = (const float*)d_in[2];   // [512]
    const float* wqkv  = (const float*)d_in[3];   // [1536,512]
    const float* wproj = (const float*)d_in[4];   // [512,512]
    const float* bproj = (const float*)d_in[5];   // [512]
    float* out = (float*)d_out;

    // workspace (bytes); AttT aliases hT (hT dead after qkv_gemm)
    char* ws = (char*)d_ws;
    _Float16* hT      = (_Float16*)(ws);                       //  8 MB  [B][N][C]
    _Float16* wqkv16  = (_Float16*)(ws + 8388608u);            //  1.5 MB
    _Float16* wproj16 = (_Float16*)(ws + 9961472u);            //  0.5 MB
    _Float16* Q16     = (_Float16*)(ws + 10485760u);           //  8 MB  [bh][n][d]
    _Float16* K16     = (_Float16*)(ws + 18874368u);           //  8 MB  [bh][n][d]
    _Float16* Vt16    = (_Float16*)(ws + 27262976u);           //  8 MB  [bh][d][n]
    _Float16* AttT    = hT;                                    //  alias [B][N][C]

    gn_kernel<<<64, 256, 0, stream>>>(x, gnw, gnb, hT);
    cvt_kernel<<<(1536 * 512 + 255) / 256, 256, 0, stream>>>(wqkv, wqkv16, 1536 * 512);
    cvt_kernel<<<(512 * 512 + 255) / 256, 256, 0, stream>>>(wproj, wproj16, 512 * 512);
    qkv_gemm<<<dim3(32, 24, 2), 128, 0, stream>>>(wqkv16, hT, Q16, K16, Vt16);
    flash_attn<<<dim3(64, 16), 128, 0, stream>>>(Q16, K16, Vt16, AttT);
    proj_gemm<<<dim3(32, 8, 2), 128, 0, stream>>>(wproj16, AttT, bproj, x, out);
}
